// LSTM_70420283785454
// MI455X (gfx1250) — compile-verified, matrix operand order checked by reading, not run on hardware
//
#include <hip/hip_runtime.h>
#include <hip/hip_bf16.h>

typedef __bf16 bf16_t;
typedef __attribute__((ext_vector_type(16))) __bf16 v16bf;
typedef __attribute__((ext_vector_type(8)))  __bf16 v8bf;
typedef __attribute__((ext_vector_type(8)))  float   v8f;
typedef __attribute__((ext_vector_type(4)))  float   v4f;

#define LSTM_B   64
#define LSTM_N   1024
#define LSTM_H   64
#define LSTM_G   256      // 4*H
#define LSTM_IN  512      // T*F

__device__ __forceinline__ float fast_sigmoid(float x) {
    return 1.0f / (1.0f + __expf(-x));
}
__device__ __forceinline__ float fast_tanh(float x) {
    float e = __expf(2.0f * x);
    return (e - 1.0f) / (e + 1.0f);
}

// ---------------------------------------------------------------------------
// Kernel 0: convert W_ih (256x512 f32, row-major) to bf16 in workspace.
// ---------------------------------------------------------------------------
__global__ void lstm_cvt_w(const float* __restrict__ w, bf16_t* __restrict__ wb, int n) {
    int i = blockIdx.x * blockDim.x + threadIdx.x;
    if (i < n) wb[i] = (bf16_t)w[i];
}

// ---------------------------------------------------------------------------
// Kernel 1: xg[r, g] = x_flat[r, :] . W_ih[g, :] + b_ih[g] + b_hh[g]
//   r = b*N + n  (natural memory order of x: (B,N,T,F) row-major)
//   65536 x 256 output, K = 512. bf16 WMMA, f32 accumulate.
//   Each wave: 16(M) x 64(N) tile, 16 K-steps of v_wmma_f32_16x16x32_bf16.
// ---------------------------------------------------------------------------
__global__ __launch_bounds__(256) void lstm_gemm_xg(
    const float*  __restrict__ x,
    const bf16_t* __restrict__ Wb,
    const float*  __restrict__ b_ih,
    const float*  __restrict__ b_hh,
    float*        __restrict__ xg)
{
    const int lane  = threadIdx.x & 31;
    const int wave  = threadIdx.x >> 5;        // 0..7
    const int mtile = blockIdx.x * 2 + (wave >> 2);   // 0..4095
    const int nquad = wave & 3;                // which 64-col group
    const int sel   = lane >> 4;               // lane half (K split)
    const int l15   = lane & 15;

    const float*  arow  = x  + (size_t)(mtile * 16 + l15) * LSTM_IN;
    const bf16_t* brow0 = Wb + (size_t)(nquad * 64 + l15) * LSTM_IN;

    v8f acc[4] = {};

    for (int k0 = 0; k0 < LSTM_IN; k0 += 32) {
        const int ka = k0 + sel * 8;
        // A fragment: row m=l15, K = {ka..ka+7} U {ka+16..ka+23}, f32->bf16
        v4f a0 = *(const v4f*)(arow + ka);
        v4f a1 = *(const v4f*)(arow + ka + 4);
        v4f a2 = *(const v4f*)(arow + ka + 16);
        v4f a3 = *(const v4f*)(arow + ka + 20);
        v16bf af;
        #pragma unroll
        for (int j = 0; j < 4; ++j) {
            af[j]      = (bf16_t)a0[j];
            af[4 + j]  = (bf16_t)a1[j];
            af[8 + j]  = (bf16_t)a2[j];
            af[12 + j] = (bf16_t)a3[j];
        }
        #pragma unroll
        for (int nt = 0; nt < 4; ++nt) {
            // B fragment = rows of W_ih (i.e. column n=l15 of B=W_ih^T)
            const bf16_t* bp = brow0 + (size_t)nt * 16 * LSTM_IN + ka;
            v8bf b0 = *(const v8bf*)(bp);
            v8bf b1 = *(const v8bf*)(bp + 16);
            v16bf bfrag;
            #pragma unroll
            for (int j = 0; j < 8; ++j) { bfrag[j] = b0[j]; bfrag[8 + j] = b1[j]; }
            acc[nt] = __builtin_amdgcn_wmma_f32_16x16x32_bf16(
                false, af, false, bfrag, (short)0, acc[nt], false, false);
        }
    }

    // Epilogue: fused bias, store f32. C/D layout: n=l15, m = v + 8*sel.
    const int rbase = mtile * 16;
    #pragma unroll
    for (int nt = 0; nt < 4; ++nt) {
        const int g   = nquad * 64 + nt * 16 + l15;
        const float bias = b_ih[g] + b_hh[g];
        #pragma unroll
        for (int v = 0; v < 8; ++v) {
            const int m = v + 8 * sel;
            xg[(size_t)(rbase + m) * LSTM_G + g] = acc[nt][v] + bias;
        }
    }
}

// ---------------------------------------------------------------------------
// Kernel 2: recurrent scan. 4 blocks (16 batch rows each) x 512 threads
// (16 waves). Wave w owns gate columns [16w, 16w+16). Per step:
//   acc <- xg tile ; acc += h @ W_hh^T (2x bf16 WMMA, W_hh frags in regs)
//   gates -> LDS ; barrier ; elementwise LSTM cell (c in registers) ;
//   h -> LDS (bf16, padded stride 72) + out ; barrier.
// ---------------------------------------------------------------------------
__global__ __launch_bounds__(512) void lstm_scan_kernel(
    const float* __restrict__ xg,
    const float* __restrict__ W_hh,
    float*       __restrict__ out)
{
    __shared__ bf16_t lds_h[16 * 72];    // 16 x 64 hidden, stride 72 (pad)
    __shared__ float  lds_g[16 * 260];   // 16 x 256 gates, stride 260 (pad)

    const int t    = threadIdx.x;
    const int lane = t & 31;
    const int w    = t >> 5;             // wave 0..15
    const int sel  = lane >> 4;
    const int l15  = lane & 15;
    const int b0   = blockIdx.x * 16;    // batch strip

    // Preload W_hh fragments (constant across all 1024 steps).
    // B = W_hh^T (K=h, N=g) => load rows of W_hh (256x64 row-major).
    v16bf bh[2];
    {
        const float* wr = W_hh + (w * 16 + l15) * LSTM_H;
        #pragma unroll
        for (int kt = 0; kt < 2; ++kt) {
            const int ka = kt * 32 + sel * 8;
            v4f c0 = *(const v4f*)(wr + ka);
            v4f c1 = *(const v4f*)(wr + ka + 4);
            v4f c2 = *(const v4f*)(wr + ka + 16);
            v4f c3 = *(const v4f*)(wr + ka + 20);
            #pragma unroll
            for (int j = 0; j < 4; ++j) {
                bh[kt][j]      = (bf16_t)c0[j];
                bh[kt][4 + j]  = (bf16_t)c1[j];
                bh[kt][8 + j]  = (bf16_t)c2[j];
                bh[kt][12 + j] = (bf16_t)c3[j];
            }
        }
    }

    // Per-lane xg base pointers for the 8 accumulator rows of this tile.
    const float* xbase[8];
    #pragma unroll
    for (int v = 0; v < 8; ++v) {
        const int m = v + 8 * sel;
        xbase[v] = xg + (size_t)(b0 + m) * LSTM_N * LSTM_G + w * 16 + l15;
    }

    // h0 = 0
    for (int i = t; i < 16 * 72; i += 512) lds_h[i] = (bf16_t)0.0f;

    // Each thread owns 2 (m,h) cells; c kept in registers across the scan.
    const int e0 = t, e1 = t + 512;
    const int m0 = e0 >> 6, h0 = e0 & 63;
    const int m1 = e1 >> 6, h1 = e1 & 63;
    float cst0 = 0.0f, cst1 = 0.0f;
    float* out0 = out + ((size_t)(b0 + m0) * LSTM_N) * LSTM_H + h0;
    float* out1 = out + ((size_t)(b0 + m1) * LSTM_N) * LSTM_H + h1;

    __syncthreads();

    for (int n = 0; n < LSTM_N; ++n) {
        // C operand initialized from the precomputed input projection.
        v8f acc;
        #pragma unroll
        for (int v = 0; v < 8; ++v) acc[v] = xbase[v][(size_t)n * LSTM_G];
        #pragma unroll
        for (int v = 0; v < 8; ++v)
            __builtin_prefetch(xbase[v] + (size_t)(n + 1) * LSTM_G, 0, 0);

        // A fragments from hidden state in LDS (row m=l15, padded stride 72).
        v16bf af0, af1;
        {
            const bf16_t* hr = lds_h + l15 * 72 + sel * 8;
            v8bf p0 = *(const v8bf*)(hr);
            v8bf q0 = *(const v8bf*)(hr + 16);
            v8bf p1 = *(const v8bf*)(hr + 32);
            v8bf q1 = *(const v8bf*)(hr + 48);
            #pragma unroll
            for (int j = 0; j < 8; ++j) {
                af0[j] = p0[j]; af0[8 + j] = q0[j];
                af1[j] = p1[j]; af1[8 + j] = q1[j];
            }
        }
        acc = __builtin_amdgcn_wmma_f32_16x16x32_bf16(
            false, af0, false, bh[0], (short)0, acc, false, false);
        acc = __builtin_amdgcn_wmma_f32_16x16x32_bf16(
            false, af1, false, bh[1], (short)0, acc, false, false);

        // Gate tile -> LDS.
        #pragma unroll
        for (int v = 0; v < 8; ++v)
            lds_g[(v + 8 * sel) * 260 + w * 16 + l15] = acc[v];
        __syncthreads();

        // Elementwise LSTM cell update (gate order: i, f, g, o).
        {
            float ig = fast_sigmoid(lds_g[m0 * 260 +   0 + h0]);
            float fg = fast_sigmoid(lds_g[m0 * 260 +  64 + h0]);
            float gg = fast_tanh   (lds_g[m0 * 260 + 128 + h0]);
            float og = fast_sigmoid(lds_g[m0 * 260 + 192 + h0]);
            cst0 = fg * cst0 + ig * gg;
            float hn = og * fast_tanh(cst0);
            lds_h[m0 * 72 + h0] = (bf16_t)hn;
            out0[(size_t)n * LSTM_H] = hn;
        }
        {
            float ig = fast_sigmoid(lds_g[m1 * 260 +   0 + h1]);
            float fg = fast_sigmoid(lds_g[m1 * 260 +  64 + h1]);
            float gg = fast_tanh   (lds_g[m1 * 260 + 128 + h1]);
            float og = fast_sigmoid(lds_g[m1 * 260 + 192 + h1]);
            cst1 = fg * cst1 + ig * gg;
            float hn = og * fast_tanh(cst1);
            lds_h[m1 * 72 + h1] = (bf16_t)hn;
            out1[(size_t)n * LSTM_H] = hn;
        }
        __syncthreads();
    }
}

// ---------------------------------------------------------------------------
extern "C" void kernel_launch(void* const* d_in, const int* in_sizes, int n_in,
                              void* d_out, int out_size, void* d_ws, size_t ws_size,
                              hipStream_t stream) {
    const float* x    = (const float*)d_in[0];  // (B,N,T,F) f32
    const float* W_ih = (const float*)d_in[1];  // (256,512) f32
    const float* W_hh = (const float*)d_in[2];  // (256,64)  f32
    const float* b_ih = (const float*)d_in[3];  // (256,)    f32
    const float* b_hh = (const float*)d_in[4];  // (256,)    f32
    float* out = (float*)d_out;                 // (B,N,H)   f32

    // Workspace layout: [xg f32 64MiB][W_ih bf16 256KiB]
    float*  xg = (float*)d_ws;
    bf16_t* Wb = (bf16_t*)((char*)d_ws +
                 (size_t)LSTM_B * LSTM_N * LSTM_G * sizeof(float));

    const int wn = LSTM_G * LSTM_IN;  // 131072
    lstm_cvt_w<<<(wn + 255) / 256, 256, 0, stream>>>(W_ih, Wb, wn);
    lstm_gemm_xg<<<2048, 256, 0, stream>>>(x, Wb, b_ih, b_hh, xg);
    lstm_scan_kernel<<<4, 512, 0, stream>>>(xg, W_hh, out);
}
